// OptitationLayer_37426345017801
// MI455X (gfx1250) — matrix operand order repaired and checked by reading, not compiled, not run on hardware
//
#include <hip/hip_runtime.h>
#include <hip/hip_bf16.h>

// ---------------------------------------------------------------------------
// Sliding-window attention layer for MI455X (gfx1250, wave32, WMMA).
//   Q = x@Wq+bq; K = x@Wk+bk; V = x@Wv+bv   (bf16 WMMA GEMMs, fp32 accum)
//   windowed softmax attention (W=9 keys), fp32 math
//   out = ctx@Wo + bo                        (bf16 WMMA GEMM, fp32 out)
// S=32768, D=1024, R=256 — all tile sizes divide exactly.
// GEMM mainloop: double-buffered LDS + async global->LDS DMA (ASYNCcnt).
// ---------------------------------------------------------------------------

typedef __attribute__((ext_vector_type(16))) __bf16 v16bf;
typedef __attribute__((ext_vector_type(8)))  __bf16 v8bf;
typedef __attribute__((ext_vector_type(4)))  __bf16 v4bf;
typedef __attribute__((ext_vector_type(8)))  float  v8f;

__device__ __forceinline__ __bf16 f2bf(float f) {
    unsigned u = __builtin_bit_cast(unsigned, f);
    unsigned r = u + 0x7fffu + ((u >> 16) & 1u);   // round-to-nearest-even
    unsigned short h = (unsigned short)(r >> 16);
    return __builtin_bit_cast(__bf16, h);
}

// Async global->LDS 16B copy (CDNA5). LDS offset = low 32 bits of generic ptr
// (ISA 10.2: LDS aperture address maps via addr[31:0]).
__device__ __forceinline__ void swa_async_b128(void* lds, const void* g) {
    unsigned l = (unsigned)(size_t)lds;
    asm volatile("global_load_async_to_lds_b128 %0, %1, off"
                 :: "v"(l), "v"(g) : "memory");
}
__device__ __forceinline__ void swa_wait_async() {
    asm volatile("s_wait_asynccnt 0x0" ::: "memory");
}

// ---------------------------------------------------------------------------
// Phase 0a: fp32 -> bf16 streaming convert (x)
// ---------------------------------------------------------------------------
__global__ __launch_bounds__(256) void swa_cvt_bf16(const float* __restrict__ in,
                                                    __bf16* __restrict__ out, int n) {
    int i = (blockIdx.x * 256 + threadIdx.x) * 4;
    if (i + 3 < n) {
        float4 f = *(const float4*)(in + i);
        v4bf o;
        o[0] = f2bf(f.x); o[1] = f2bf(f.y); o[2] = f2bf(f.z); o[3] = f2bf(f.w);
        *(v4bf*)(out + i) = o;
    }
}

// ---------------------------------------------------------------------------
// Phase 0b: fp32 [Kd][Nd] -> bf16 transposed [Nd][Kd]  (weights; tiny)
// ---------------------------------------------------------------------------
__global__ __launch_bounds__(256) void swa_transpose_cvt(const float* __restrict__ in,
                                                         __bf16* __restrict__ out,
                                                         int Kd, int Nd) {
    int idx = blockIdx.x * 256 + threadIdx.x;
    if (idx < Kd * Nd) {
        int k = idx / Nd, n = idx % Nd;
        out[(size_t)n * Kd + k] = f2bf(in[idx]);
    }
}

// ---------------------------------------------------------------------------
// WMMA GEMM:  C[M][N] = A[M][K] * Bt[N][K]^T + bias[N]
//   A, Bt bf16 row-major (Bt is B transposed so B fragments are K-contiguous).
//   Workgroup: 256 threads = 8 waves; tile 128(M) x 128(N).
//   Wave grid 2(M) x 4(N): wave tile 64x32 = 4x2 fragments of 16x16.
//   Double-buffered LDS tiles (stride 40 elems = 80B, conflict-free b128
//   fragment reads); async DMA for tile t+1 overlaps WMMA on tile t.
// ---------------------------------------------------------------------------
#define LDST 40

__device__ __forceinline__ void swa_store_out(float* p, float v)  { *p = v; }
__device__ __forceinline__ void swa_store_out(__bf16* p, float v) { *p = f2bf(v); }

template <typename OUT_T>
__global__ __launch_bounds__(256) void swa_wmma_gemm(const __bf16* __restrict__ A,
                                                     const __bf16* __restrict__ Bt,
                                                     const float* __restrict__ bias,
                                                     OUT_T* __restrict__ C,
                                                     int M, int N, int K) {
    __shared__ __bf16 As[2][128 * LDST];
    __shared__ __bf16 Bs[2][128 * LDST];

    const int m0 = blockIdx.x * 128;
    const int n0 = blockIdx.y * 128;
    const int tid  = threadIdx.x;
    const int wave = tid >> 5;
    const int lane = tid & 31;
    const int wm = wave >> 2;        // 0..1 : which 64-row half
    const int wn = wave & 3;         // 0..3 : which 32-col quarter
    const int lr = lane & 15;        // fragment row/col within 16
    const int lh = lane >> 4;        // lane half selects K sub-blocks

    // Staging assignment: 512 16B chunks per tile, 2 per thread (no branches).
    const int row0 = tid >> 2,          seg0 = tid & 3;          // rows 0..63
    const int row1 = (tid + 256) >> 2,  seg1 = tid & 3;          // rows 64..127

    v8f acc[4][2];
#pragma unroll
    for (int i = 0; i < 4; ++i)
#pragma unroll
        for (int j = 0; j < 2; ++j)
#pragma unroll
            for (int r = 0; r < 8; ++r) acc[i][j][r] = 0.0f;

    auto issue = [&](int buf, int kk) {
        swa_async_b128(&As[buf][row0 * LDST + seg0 * 8],
                       &A[(size_t)(m0 + row0) * K + kk + seg0 * 8]);
        swa_async_b128(&As[buf][row1 * LDST + seg1 * 8],
                       &A[(size_t)(m0 + row1) * K + kk + seg1 * 8]);
        swa_async_b128(&Bs[buf][row0 * LDST + seg0 * 8],
                       &Bt[(size_t)(n0 + row0) * K + kk + seg0 * 8]);
        swa_async_b128(&Bs[buf][row1 * LDST + seg1 * 8],
                       &Bt[(size_t)(n0 + row1) * K + kk + seg1 * 8]);
    };

    issue(0, 0);
    int buf = 0;
    for (int kk = 0; kk < K; kk += 32) {
        swa_wait_async();      // our batch for `buf` has landed in LDS
        __syncthreads();       // everyone's batch has landed; prev reads done
        if (kk + 32 < K) issue(buf ^ 1, kk + 32);

        // ---- A fragments (lane half selects K block 0-15 / 16-31, matching B)
        v16bf a[4];
#pragma unroll
        for (int i = 0; i < 4; ++i) {
            const __bf16* p = &As[buf][(wm * 64 + i * 16 + lr) * LDST + lh * 16];
            v8bf lo = *(const v8bf*)p;          // K = lh*16 .. +7
            v8bf hi = *(const v8bf*)(p + 8);    // K = lh*16+8 .. +15
            a[i] = __builtin_shufflevector(lo, hi, 0, 1, 2, 3, 4, 5, 6, 7,
                                           8, 9, 10, 11, 12, 13, 14, 15);
        }
        // ---- B fragments (lane half selects K block 0-15 / 16-31, matching A)
        v16bf b[2];
#pragma unroll
        for (int j = 0; j < 2; ++j) {
            const __bf16* p = &Bs[buf][(wn * 32 + j * 16 + lr) * LDST + lh * 16];
            v8bf lo = *(const v8bf*)p;          // K = lh*16 .. +7
            v8bf hi = *(const v8bf*)(p + 8);    // K = lh*16+8 .. +15
            b[j] = __builtin_shufflevector(lo, hi, 0, 1, 2, 3, 4, 5, 6, 7,
                                           8, 9, 10, 11, 12, 13, 14, 15);
        }
        // ---- 8 WMMAs: 64x32 wave tile, K += 32
#pragma unroll
        for (int i = 0; i < 4; ++i)
#pragma unroll
            for (int j = 0; j < 2; ++j)
                acc[i][j] = __builtin_amdgcn_wmma_f32_16x16x32_bf16(
                    false, a[i], false, b[j], (short)0, acc[i][j], false, false);
        buf ^= 1;
    }

    // ---- epilogue: C layout VGPR r -> M = r + 8*laneHalf, lane%16 -> N
#pragma unroll
    for (int i = 0; i < 4; ++i)
#pragma unroll
        for (int j = 0; j < 2; ++j) {
            const int n = n0 + wn * 32 + j * 16 + lr;
            const float bn = bias[n];
#pragma unroll
            for (int r = 0; r < 8; ++r) {
                const int m = m0 + wm * 64 + i * 16 + r + 8 * lh;
                swa_store_out(&C[(size_t)m * N + n], acc[i][j][r] + bn);
            }
        }
}

// ---------------------------------------------------------------------------
// Phase 2: sliding-window attention. R=256 fixed, window W=9.
// Block = 256 threads = 8 waves = 8 consecutive queries.
// LDS stages the 16 K rows and 16 V rows the block needs (fp32).
// One wave per query; lane owns 8 of the 256 channels.
// ---------------------------------------------------------------------------
__global__ __launch_bounds__(256) void swa_attn(const __bf16* __restrict__ Qb,
                                                const __bf16* __restrict__ Kb,
                                                const __bf16* __restrict__ Vb,
                                                __bf16* __restrict__ Ctx, int S) {
    __shared__ float Ks[16][256];
    __shared__ float Vs[16][256];

    const int s0   = blockIdx.x * 8;
    const int tid  = threadIdx.x;
    const int wave = tid >> 5;
    const int lane = tid & 31;

    // stage K/V rows s0-8 .. s0+7 (clamped; invalid ones are masked later)
    {
        const int row  = tid >> 4;            // 0..15
        const int col0 = (tid & 15) * 16;     // 0..240
        int g = s0 - 8 + row; if (g < 0) g = 0;
        const uint4* kp = (const uint4*)(Kb + (size_t)g * 256 + col0);
        const uint4* vp = (const uint4*)(Vb + (size_t)g * 256 + col0);
        uint4 k0 = kp[0], k1 = kp[1], v0 = vp[0], v1 = vp[1];
        unsigned kw[8] = {k0.x, k0.y, k0.z, k0.w, k1.x, k1.y, k1.z, k1.w};
        unsigned vw[8] = {v0.x, v0.y, v0.z, v0.w, v1.x, v1.y, v1.z, v1.w};
#pragma unroll
        for (int e = 0; e < 8; ++e) {
            Ks[row][col0 + 2 * e]     = __builtin_bit_cast(float, kw[e] << 16);
            Ks[row][col0 + 2 * e + 1] = __builtin_bit_cast(float, kw[e] & 0xffff0000u);
            Vs[row][col0 + 2 * e]     = __builtin_bit_cast(float, vw[e] << 16);
            Vs[row][col0 + 2 * e + 1] = __builtin_bit_cast(float, vw[e] & 0xffff0000u);
        }
    }
    __syncthreads();

    const int s = s0 + wave;
    // query row, pre-scaled by 1/sqrt(R) = 1/16
    float q[8];
    {
        const uint4* qp = (const uint4*)(Qb + (size_t)s * 256 + lane * 8);
        uint4 q0 = qp[0];
        unsigned qw[4] = {q0.x, q0.y, q0.z, q0.w};
#pragma unroll
        for (int e = 0; e < 4; ++e) {
            q[2 * e]     = __builtin_bit_cast(float, qw[e] << 16) * 0.0625f;
            q[2 * e + 1] = __builtin_bit_cast(float, qw[e] & 0xffff0000u) * 0.0625f;
        }
    }

    float sc[9];
#pragma unroll
    for (int w = 0; w < 9; ++w) {
        const float* kr = &Ks[wave + w][lane * 8];
        float p = 0.0f;
#pragma unroll
        for (int e = 0; e < 8; ++e) p = fmaf(q[e], kr[e], p);
#pragma unroll
        for (int m = 16; m >= 1; m >>= 1) p += __shfl_xor(p, m, 32);
        sc[w] = (s - 8 + w >= 0) ? p : -__builtin_inff();
    }

    float mx = sc[0];
#pragma unroll
    for (int w = 1; w < 9; ++w) mx = fmaxf(mx, sc[w]);
    float aw[9], se = 0.0f;
#pragma unroll
    for (int w = 0; w < 9; ++w) { aw[w] = __expf(sc[w] - mx); se += aw[w]; }
    const float inv = 1.0f / se;

    float o[8];
#pragma unroll
    for (int e = 0; e < 8; ++e) o[e] = 0.0f;
#pragma unroll
    for (int w = 0; w < 9; ++w) {
        const float a = aw[w] * inv;
        const float* vr = &Vs[wave + w][lane * 8];
#pragma unroll
        for (int e = 0; e < 8; ++e) o[e] = fmaf(a, vr[e], o[e]);
    }
    v8bf ob;
#pragma unroll
    for (int e = 0; e < 8; ++e) ob[e] = f2bf(o[e]);
    *(v8bf*)(Ctx + (size_t)s * 256 + lane * 8) = ob;
}

// ---------------------------------------------------------------------------
// Host launch. Workspace layout (bytes, 256-aligned, total ~131 MB):
//   xb 64MB | WqT/WkT/WvT/WoT 512KB each | Qb/Kb/Vb/Ctxb 16MB each
// ---------------------------------------------------------------------------
extern "C" void kernel_launch(void* const* d_in, const int* in_sizes, int n_in,
                              void* d_out, int out_size, void* d_ws, size_t ws_size,
                              hipStream_t stream) {
    (void)in_sizes; (void)n_in; (void)out_size; (void)ws_size;
    constexpr int S = 32768, D = 1024, R = 256;

    const float* x  = (const float*)d_in[0];
    const float* Wq = (const float*)d_in[1];
    const float* Wk = (const float*)d_in[2];
    const float* Wv = (const float*)d_in[3];
    const float* Wo = (const float*)d_in[4];
    const float* bq = (const float*)d_in[5];
    const float* bk = (const float*)d_in[6];
    const float* bv = (const float*)d_in[7];
    const float* bo = (const float*)d_in[8];

    char* ws = (char*)d_ws;
    auto take = [&](size_t bytes) { char* p = ws; ws += (bytes + 255) & ~size_t(255); return p; };
    __bf16* xb   = (__bf16*)take((size_t)S * D * 2);
    __bf16* WqT  = (__bf16*)take((size_t)R * D * 2);
    __bf16* WkT  = (__bf16*)take((size_t)R * D * 2);
    __bf16* WvT  = (__bf16*)take((size_t)R * D * 2);
    __bf16* WoT  = (__bf16*)take((size_t)D * R * 2);
    __bf16* Qb   = (__bf16*)take((size_t)S * R * 2);
    __bf16* Kbuf = (__bf16*)take((size_t)S * R * 2);
    __bf16* Vbuf = (__bf16*)take((size_t)S * R * 2);
    __bf16* Ctxb = (__bf16*)take((size_t)S * R * 2);

    // Phase 0: precision conversion (x streaming; weights transposed)
    swa_cvt_bf16<<<(S * D) / (256 * 4), 256, 0, stream>>>(x, xb, S * D);
    swa_transpose_cvt<<<(D * R) / 256, 256, 0, stream>>>(Wq, WqT, D, R);
    swa_transpose_cvt<<<(D * R) / 256, 256, 0, stream>>>(Wk, WkT, D, R);
    swa_transpose_cvt<<<(D * R) / 256, 256, 0, stream>>>(Wv, WvT, D, R);
    swa_transpose_cvt<<<(R * D) / 256, 256, 0, stream>>>(Wo, WoT, R, D);

    // Phase 1: QKV projections (bf16 WMMA, fp32 accum)
    dim3 g1(S / 128, R / 128);
    swa_wmma_gemm<__bf16><<<g1, 256, 0, stream>>>(xb, WqT, bq, Qb, S, R, D);
    swa_wmma_gemm<__bf16><<<g1, 256, 0, stream>>>(xb, WkT, bk, Kbuf, S, R, D);
    swa_wmma_gemm<__bf16><<<g1, 256, 0, stream>>>(xb, WvT, bv, Vbuf, S, R, D);

    // Phase 2: sliding-window softmax attention
    swa_attn<<<S / 8, 256, 0, stream>>>(Qb, Kbuf, Vbuf, Ctxb, S);

    // Phase 3: output projection -> fp32 d_out
    dim3 g2(S / 128, D / 128);
    swa_wmma_gemm<float><<<g2, 256, 0, stream>>>(Ctxb, WoT, bo, (float*)d_out, S, D, R);
}
